// GCN_38972533244082
// MI455X (gfx1250) — matrix operand order repaired and checked by reading, not compiled, not run on hardware
//
#include <hip/hip_runtime.h>

#define D_IN 256
#define D    128

typedef __attribute__((ext_vector_type(16))) __bf16 vbf16;
typedef __attribute__((ext_vector_type(8)))  __bf16 vbf8;
typedef __attribute__((ext_vector_type(4)))  __bf16 vbf4;
typedef __attribute__((ext_vector_type(8)))  float  v8f;

// ---------------- WMMA helpers (CDNA5 bf16 16x16x32, f32 accum) ----------------

static __device__ __forceinline__ v8f wmma_bf16(vbf16 a, vbf16 b, v8f c) {
  // 8 args: (neg_a, A, neg_b, B, c_mod, C, reuse_a, reuse_b)
  return __builtin_amdgcn_wmma_f32_16x16x32_bf16(false, a, false, b, (short)0, c,
                                                 false, false);
}

// A (16xK row-major in LDS) -> per-lane vbf16 in contiguous k order (k = 16*(lane/16) + s), same as B:
// lane<16: K[0..15]; lane>=16: K[16..31]
static __device__ __forceinline__ vbf16 load_A_lds(const __bf16* As, int K,
                                                   int kblock, int lane) {
  int m  = lane & 15;
  int kb = kblock * 32 + ((lane >> 4) << 4);
  const __bf16* p = As + m * K + kb;
  vbf8 lo = *(const vbf8*)(p);        // K = kb .. kb+7   (ds_load_b128)
  vbf8 hi = *(const vbf8*)(p + 8);    // K = kb+8 .. kb+15
  return __builtin_shufflevector(lo, hi, 0, 1, 2, 3, 4, 5, 6, 7,
                                 8, 9, 10, 11, 12, 13, 14, 15);
}

// B (KxN) = W^T where W is [Dout,K] row-major bf16.
// Contiguous k order (same as load_A_lds): lane<16 holds col n=lane, K[0..15];
// lane>=16 holds col n=lane-16, K[16..31] -> one contiguous 32B run of W's row n.
static __device__ __forceinline__ vbf16 load_B_row(const __bf16* __restrict__ W,
                                                   int K, int ntile, int kblock,
                                                   int lane) {
  int n  = ntile * 16 + (lane & 15);
  int kb = kblock * 32 + ((lane >> 4) << 4);
  return *(const vbf16*)(W + (size_t)n * K + kb);
}

// ---------------- small utility kernels ----------------

__global__ void k_fill(float* __restrict__ p, float v, int n) {
  int i = blockIdx.x * blockDim.x + threadIdx.x;
  if (i < n) p[i] = v;
}

__global__ void k_zero4(float4* __restrict__ p, long n4) {
  long i = (long)blockIdx.x * blockDim.x + threadIdx.x;
  if (i < n4) p[i] = make_float4(0.f, 0.f, 0.f, 0.f);
}

__global__ void k_deg(const int* __restrict__ dst, float* __restrict__ deg, int E) {
  int e = blockIdx.x * blockDim.x + threadIdx.x;
  if (e < E) atomicAdd(&deg[dst[e]], 1.0f);
}

__global__ void k_rsqrt(float* __restrict__ d, int n) {
  int i = blockIdx.x * blockDim.x + threadIdx.x;
  if (i < n) d[i] = rsqrtf(d[i]);   // deg >= 1 (self loop) so no clamp needed
}

__global__ void k_cvt(const float* __restrict__ in, __bf16* __restrict__ out, int n) {
  int i = blockIdx.x * blockDim.x + threadIdx.x;
  if (i < n) out[i] = (__bf16)in[i];
}

// ---------------- fused embed + conv1 GEMM ----------------
// h1[16 nodes, D] = (embed[token] @ Wn^T + bn) @ W1^T ; x tile staged via LDS

__global__ void __launch_bounds__(256)
k_fused_l1(const int* __restrict__ tokens, const float* __restrict__ emb,
           const __bf16* __restrict__ Wn, const float* __restrict__ bias,
           const __bf16* __restrict__ W1, __bf16* __restrict__ h1out, int N) {
  __shared__ __attribute__((aligned(16))) __bf16 As[16 * D_IN];  // 8 KB
  __shared__ __attribute__((aligned(16))) __bf16 Xs[16 * D];     // 4 KB
  int mt = blockIdx.x;
  int t  = threadIdx.x;               // 256 threads = 8 waves
  // stage A: gather 16 embedding rows, f32 -> bf16, fully coalesced per row
  for (int j = 0; j < 16; ++j) {
    int row = mt * 16 + j;
    int tok = tokens[row < N ? row : N - 1];   // block-uniform -> scalar load
    As[j * D_IN + t] = (__bf16)emb[(size_t)tok * D_IN + t];
  }
  __syncthreads();

  int lane = t & 31;
  int wv   = t >> 5;                  // 8 waves -> 8 column tiles of 16
  v8f c = {};
  for (int kb = 0; kb < D_IN / 32; ++kb) {
    vbf16 a = load_A_lds(As, D_IN, kb, lane);
    vbf16 b = load_B_row(Wn, D_IN, wv, kb, lane);
    c = wmma_bf16(a, b, c);
  }
  // x tile (with bias) -> LDS as bf16; C layout: VGPR r -> M = r + 8*(lane>>4)
  union { v8f v; float f[8]; } cu; cu.v = c;
  int ncol = wv * 16 + (lane & 15);
  float bn = bias[ncol];
  int mloc = ((lane >> 4) << 3);
  for (int r = 0; r < 8; ++r)
    Xs[(mloc + r) * D + ncol] = (__bf16)(cu.f[r] + bn);
  __syncthreads();

  // second GEMM: h1 = x @ W1^T (no bias; bias applied after aggregation)
  v8f c2 = {};
  for (int kb = 0; kb < D / 32; ++kb) {
    vbf16 a = load_A_lds(Xs, D, kb, lane);
    vbf16 b = load_B_row(W1, D, wv, kb, lane);
    c2 = wmma_bf16(a, b, c2);
  }
  union { v8f v; float f[8]; } du; du.v = c2;
  int mbase = mt * 16 + mloc;
  for (int r = 0; r < 8; ++r) {
    int row = mbase + r;
    if (row < N) h1out[(size_t)row * D + ncol] = (__bf16)du.f[r];
  }
}

// ---------------- conv GEMM with async global->LDS A staging ----------------
// h[16 nodes, D] = x @ W^T ; A tile (4 KB) staged via GLOBAL_LOAD_ASYNC_TO_LDS_B128

__global__ void __launch_bounds__(256)
k_conv_gemm(const __bf16* __restrict__ xin, const __bf16* __restrict__ W,
            __bf16* __restrict__ hout, int N) {
  __shared__ __attribute__((aligned(16))) __bf16 As[16 * D];     // 4096 B
  int mt = blockIdx.x;
  int t  = threadIdx.x;
  // 256 threads x 16 B = exactly the 16x128 bf16 tile; ASYNCcnt path, no VGPR bounce
  {
    unsigned lds_addr = (unsigned)(uintptr_t)(&As[0]) + (unsigned)t * 16u;
    unsigned long long gaddr =
        (unsigned long long)(uintptr_t)(xin + (size_t)mt * 16 * D) +
        (unsigned long long)t * 16ull;
    asm volatile("global_load_async_to_lds_b128 %0, %1, off"
                 :: "v"(lds_addr), "v"(gaddr) : "memory");
    asm volatile("s_wait_asynccnt 0x0" ::: "memory");
  }
  __syncthreads();

  int lane = t & 31;
  int wv   = t >> 5;
  v8f c = {};
  for (int kb = 0; kb < D / 32; ++kb) {
    vbf16 a = load_A_lds(As, D, kb, lane);
    vbf16 b = load_B_row(W, D, wv, kb, lane);
    c = wmma_bf16(a, b, c);
  }
  union { v8f v; float f[8]; } cu; cu.v = c;
  int ncol  = wv * 16 + (lane & 15);
  int mbase = mt * 16 + ((lane >> 4) << 3);
  for (int r = 0; r < 8; ++r) {
    int row = mbase + r;
    if (row < N) hout[(size_t)row * D + ncol] = (__bf16)cu.f[r];
  }
}

// ---------------- edge scatter: acc[dst] += h[src] * dinv[src]*dinv[dst] ----------------
// one wave per edge; each lane owns 4 consecutive features (8B bf16 load,
// 4x global_atomic_add_f32, coalesced 512B per edge)
__global__ void __launch_bounds__(256)
k_scatter(const int* __restrict__ src, const int* __restrict__ dst,
          const float* __restrict__ dinv, const __bf16* __restrict__ h,
          float* __restrict__ acc, int E) {
  int gid  = blockIdx.x * blockDim.x + threadIdx.x;
  int e    = gid >> 5;
  int lane = threadIdx.x & 31;
  if (e >= E) return;
  e = __builtin_amdgcn_readfirstlane(e);       // wave-uniform -> scalar loads
  int s = src[e];
  int d = dst[e];
  float coef = dinv[s] * dinv[d];
  vbf4 hv = *(const vbf4*)(h + (size_t)s * D + lane * 4);
  float* ap = acc + (size_t)d * D + lane * 4;
  atomicAdd(ap + 0, (float)hv[0] * coef);
  atomicAdd(ap + 1, (float)hv[1] * coef);
  atomicAdd(ap + 2, (float)hv[2] * coef);
  atomicAdd(ap + 3, (float)hv[3] * coef);
}

// epilogue layer1: add self-loop message + bias, ReLU, emit bf16
__global__ void k_post_relu(const float* __restrict__ acc, const __bf16* __restrict__ h1,
                            const float* __restrict__ dinv, const float* __restrict__ b,
                            __bf16* __restrict__ hout, long total) {
  long i = (long)blockIdx.x * blockDim.x + threadIdx.x;
  if (i >= total) return;
  int node = (int)(i >> 7), col = (int)(i & (D - 1));
  float di = dinv[node];
  float v = acc[i] + (float)h1[i] * di * di + b[col];
  hout[i] = (__bf16)fmaxf(v, 0.f);
}

// epilogue layer2 (in place on d_out): out += self-loop message + bias
__global__ void k_post_out(float* __restrict__ out, const __bf16* __restrict__ h2,
                           const float* __restrict__ dinv, const float* __restrict__ b,
                           long total) {
  long i = (long)blockIdx.x * blockDim.x + threadIdx.x;
  if (i >= total) return;
  int node = (int)(i >> 7), col = (int)(i & (D - 1));
  float di = dinv[node];
  out[i] = out[i] + (float)h2[i] * di * di + b[col];
}

// ---------------- launch ----------------

extern "C" void kernel_launch(void* const* d_in, const int* in_sizes, int n_in,
                              void* d_out, int out_size, void* d_ws, size_t ws_size,
                              hipStream_t stream) {
  const int*   tokens = (const int*)d_in[0];
  const int*   eidx   = (const int*)d_in[1];
  const float* emb    = (const float*)d_in[2];
  const float* Wn     = (const float*)d_in[3];
  const float* bn     = (const float*)d_in[4];
  const float* W1     = (const float*)d_in[5];
  const float* b1     = (const float*)d_in[6];
  const float* W2     = (const float*)d_in[7];
  const float* b2     = (const float*)d_in[8];
  int N = in_sizes[0];
  int E = in_sizes[1] / 2;
  const int* esrc = eidx;
  const int* edst = eidx + E;

  char* ws = (char*)d_ws;
  size_t off = 0;
  auto alloc = [&](size_t bytes) -> char* {
    char* p = ws + off;
    off += (bytes + 255) & ~(size_t)255;
    return p;
  };
  float*  dinv  = (float*)alloc((size_t)N * 4);
  __bf16* Wn_bf = (__bf16*)alloc((size_t)D * D_IN * 2);
  __bf16* W1_bf = (__bf16*)alloc((size_t)D * D * 2);
  __bf16* W2_bf = (__bf16*)alloc((size_t)D * D * 2);
  __bf16* xbuf  = (__bf16*)alloc((size_t)N * D * 2);  // relu(h) between layers
  __bf16* hbuf  = (__bf16*)alloc((size_t)N * D * 2);  // h1, then h2
  float*  acc   = (float*)alloc((size_t)N * D * 4);
  float*  outf  = (float*)d_out;

  int  mtiles = (N + 15) / 16;
  long ND     = (long)N * D;
  int  gN     = (N + 255) / 256;
  int  gE     = (E + 255) / 256;
  int  gE32   = (int)(((long)E * 32 + 255) / 256);
  int  gND    = (int)((ND + 255) / 256);
  int  gND4   = (int)((ND / 4 + 255) / 256);

  // degree + symmetric normalization (deg starts at 1 for the self loop)
  k_fill<<<gN, 256, 0, stream>>>(dinv, 1.0f, N);
  k_deg<<<gE, 256, 0, stream>>>(edst, dinv, E);
  k_rsqrt<<<gN, 256, 0, stream>>>(dinv, N);

  // weights -> bf16 (once per launch, tiny)
  k_cvt<<<(D * D_IN + 255) / 256, 256, 0, stream>>>(Wn, Wn_bf, D * D_IN);
  k_cvt<<<(D * D + 255) / 256, 256, 0, stream>>>(W1, W1_bf, D * D);
  k_cvt<<<(D * D + 255) / 256, 256, 0, stream>>>(W2, W2_bf, D * D);

  // layer 1 (fused): h1 = (embed[tokens] @ Wn^T + bn) @ W1^T ; scatter ; relu epilogue
  k_zero4<<<gND4, 256, 0, stream>>>((float4*)acc, ND / 4);
  k_fused_l1<<<mtiles, 256, 0, stream>>>(tokens, emb, Wn_bf, bn, W1_bf, hbuf, N);
  k_scatter<<<gE32, 256, 0, stream>>>(esrc, edst, dinv, hbuf, acc, E);
  k_post_relu<<<gND, 256, 0, stream>>>(acc, hbuf, dinv, b1, xbuf, ND);

  // layer 2: h2 = h @ W2^T ; scatter straight into d_out ; +bias+self loop
  k_zero4<<<gND4, 256, 0, stream>>>((float4*)outf, ND / 4);
  k_conv_gemm<<<mtiles, 256, 0, stream>>>(xbuf, W2_bf, hbuf, N);
  k_scatter<<<gE32, 256, 0, stream>>>(esrc, edst, dinv, hbuf, outf, E);
  k_post_out<<<gND, 256, 0, stream>>>(outf, hbuf, dinv, b2, ND);
}
